// VectorQuantizer_70231305224702
// MI455X (gfx1250) — compile-verified, matrix operand order checked by reading, not run on hardware
//
#include <hip/hip_runtime.h>
#include <hip/hip_bf16.h>

// ---------------------------------------------------------------------------
// VQ-VAE vector quantizer for MI455X (gfx1250, wave32, WMMA + async-to-LDS).
//
// z:        [8, 64, 8192] f32 ; codebook: [1024, 64] f32
// outputs (concat in d_out, f32): z_q [8,64,8192], vq_loss [1], codes [65536]
//
// Score GEMM (65536 x 1024 x 64) via V_WMMA_F32_16X16X32_BF16 with 2-term
// bf16 split (hi*Whi + lo*Whi + hi*Wlo, f32 acc) -> ~f32-accurate scores,
// 6 WMMAs / 16x16 tile in 3 independent accumulator chains (XDL ILP = 3).
// Codebook fragments are staged global->LDS with double-buffered
// GLOBAL_LOAD_ASYNC_TO_LDS_B128 (ASYNCcnt), fetched once per block.
// ---------------------------------------------------------------------------

typedef __attribute__((ext_vector_type(16))) __bf16 v16bf;
typedef __attribute__((ext_vector_type(8)))  float  v8f;

#define VQ_B   8
#define VQ_D   64
#define VQ_T   8192
#define VQ_N   (VQ_B * VQ_T)          // 65536 queries
#define VQ_CB  1024                   // codes
#define Z_ELEMS (VQ_B * VQ_D * VQ_T)  // 4194304

#define NTILES       64               // 1024 codes / 16
#define CHUNK_TILES  4
#define NCHUNKS      (NTILES / CHUNK_TILES)           // 16
#define TILE_HALVES  (4 * 32 * 16)                    // hi0,lo0,hi1,lo1 frags
#define CHUNK_HALVES (CHUNK_TILES * TILE_HALVES)      // 8192 bf16
#define CHUNK_BYTES  (CHUNK_HALVES * 2)               // 16384 B
#define COPIES_PER_WAVE 4                             // 4 * 32 lanes * 16B = 2KB

#if __has_builtin(__builtin_amdgcn_s_wait_asynccnt)
#define ASYNC_WAIT(n) __builtin_amdgcn_s_wait_asynccnt(n)
#else
#define ASYNC_WAIT(n) asm volatile("s_wait_asynccnt %0" :: "i"(n) : "memory")
#endif

// K index inside a 32-wide K-tile for 16-bit A/B fragments (ISA 7.12.2):
// VGPR v (=e>>1), half hh (=e&1); lanes 16-31 carry K+8 within each 16-K half.
__device__ __forceinline__ int frag_klocal(int e, int laneHigh) {
    int v = e >> 1, hh = e & 1;
    int kl = (v < 4) ? (2 * v + hh) : (16 + 2 * (v - 4) + hh);
    return kl + (laneHigh ? 8 : 0);
}

// --- prep: pack codebook into interleaved WMMA B fragments (bf16 hi/lo) -----
// layout: per tile: [hi kt0][lo kt0][hi kt1][lo kt1], each 32 lanes x 16 halves.
// element index: idx = ((tile*4 + f)*32 + lane)*16 + e
__global__ __launch_bounds__(256)
void vq_prep_frags(const float* __restrict__ W, __bf16* __restrict__ comb) {
    int idx  = blockIdx.x * 256 + threadIdx.x;   // 131072 total
    int e    = idx & 15;
    int lane = (idx >> 4) & 31;
    int f    = (idx >> 9) & 3;
    int tile = idx >> 11;
    int kt = f >> 1, isLo = f & 1;
    int k = kt * 32 + frag_klocal(e, lane >= 16);
    int j = tile * 16 + (lane & 15);             // code id (B column)
    float val = W[j * VQ_D + k];
    __bf16 h = (__bf16)val;
    comb[idx] = isLo ? (__bf16)(val - (float)h) : h;
}

__global__ __launch_bounds__(256)
void vq_prep_wnorm(const float* __restrict__ W, float* __restrict__ wnorm) {
    int j = blockIdx.x * 256 + threadIdx.x;
    if (j < VQ_CB) {
        float s = 0.f;
        #pragma unroll
        for (int d = 0; d < VQ_D; ++d) { float w = W[j * VQ_D + d]; s += w * w; }
        wnorm[j] = s;
    }
}

// --- main: 512 blocks x 256 threads; each wave owns 16 queries ---------------
__global__ __launch_bounds__(256)
void vq_main(const float* __restrict__ z, const float* __restrict__ W,
             const __bf16* __restrict__ comb, const float* __restrict__ wnorm,
             float* __restrict__ out_zq, float* __restrict__ out_codes,
             float* __restrict__ partials) {
    __shared__ __attribute__((aligned(32))) __bf16 bbuf[2][CHUNK_HALVES]; // 2x16KB
    __shared__ float wnorm_lds[VQ_CB];
    __shared__ int   codes_lds[128];
    __shared__ float wavesum[8];

    const int tid      = threadIdx.x;
    const int lane     = tid & 31;
    const int wave     = tid >> 5;
    const int laneHigh = lane >> 4;
    const int m16      = lane & 15;

    // -- async copy of one B-fragment chunk: each wave moves 2KB -------------
    const char* gbase = (const char*)comb;
    auto issue_chunk = [&](int c, int buf) {
        unsigned ldsbase = (unsigned)(unsigned long long)(&bbuf[buf][0]); // LDS offset in addr[31:0]
        #pragma unroll
        for (int i = 0; i < COPIES_PER_WAVE; ++i) {
            int byte = wave * (COPIES_PER_WAVE * 512) + i * 512 + lane * 16;
            unsigned ldsa = ldsbase + byte;
            unsigned long long ga =
                (unsigned long long)(gbase + (size_t)c * CHUNK_BYTES + byte);
            asm volatile("global_load_async_to_lds_b128 %0, %1, off"
                         :: "v"(ldsa), "v"(ga) : "memory");
        }
    };

    issue_chunk(0, 0);  // prefetch first chunk before doing anything else

    // ||W||^2 table -> LDS (read every tile; visibility covered by c=0 barrier)
    for (int i = tid; i < VQ_CB; i += 256) wnorm_lds[i] = wnorm[i];

    // A fragments: 16 queries of this wave, all 64 dims, split bf16 hi/lo.
    // Loads coalesced across lanes (fixed dim d => consecutive t).
    const int n     = blockIdx.x * 128 + wave * 16 + m16;
    const int zbase = (n >> 13) * (VQ_D * VQ_T) + (n & (VQ_T - 1));
    v16bf ahi[2], alo[2];
    #pragma unroll
    for (int kt = 0; kt < 2; ++kt) {
        #pragma unroll
        for (int e = 0; e < 16; ++e) {
            int d = kt * 32 + frag_klocal(e, laneHigh);
            float x = z[zbase + d * VQ_T];
            __bf16 h = (__bf16)x;
            ahi[kt][e] = h;
            alo[kt][e] = (__bf16)(x - (float)h);
        }
    }

    // Per-lane running argmin of (||W_j||^2 - 2 z.W_j); lane covers n%16==m16.
    float best[8]; int bidx[8];
    #pragma unroll
    for (int r = 0; r < 8; ++r) { best[r] = 3.4e38f; bidx[r] = 0; }

    for (int c = 0; c < NCHUNKS; ++c) {
        if (c + 1 < NCHUNKS) {
            issue_chunk(c + 1, (c + 1) & 1);
            ASYNC_WAIT(COPIES_PER_WAVE);   // chunk c complete (in-order)
        } else {
            ASYNC_WAIT(0);
        }
        __syncthreads();                   // chunk c visible to all waves

        const v16bf* pb = (const v16bf*)&bbuf[c & 1][0];
        #pragma unroll
        for (int tl = 0; tl < CHUNK_TILES; ++tl) {
            int tile = c * CHUNK_TILES + tl;
            v16bf bh0 = pb[(tl * 4 + 0) * 32 + lane];
            v16bf bl0 = pb[(tl * 4 + 1) * 32 + lane];
            v16bf bh1 = pb[(tl * 4 + 2) * 32 + lane];
            v16bf bl1 = pb[(tl * 4 + 3) * 32 + lane];
            v8f a0 = {0.f,0.f,0.f,0.f,0.f,0.f,0.f,0.f};
            v8f a1 = a0, a2 = a0;
            // 3 independent XDL chains: S = hi*Whi + lo*Whi + hi*Wlo
            a0 = __builtin_amdgcn_wmma_f32_16x16x32_bf16(false, ahi[0], false, bh0, (short)0, a0, false, false);
            a1 = __builtin_amdgcn_wmma_f32_16x16x32_bf16(false, alo[0], false, bh0, (short)0, a1, false, false);
            a2 = __builtin_amdgcn_wmma_f32_16x16x32_bf16(false, ahi[0], false, bl0, (short)0, a2, false, false);
            a0 = __builtin_amdgcn_wmma_f32_16x16x32_bf16(false, ahi[1], false, bh1, (short)0, a0, false, false);
            a1 = __builtin_amdgcn_wmma_f32_16x16x32_bf16(false, alo[1], false, bh1, (short)0, a1, false, false);
            a2 = __builtin_amdgcn_wmma_f32_16x16x32_bf16(false, ahi[1], false, bl1, (short)0, a2, false, false);
            v8f s = a0 + a1 + a2;

            float wn = wnorm_lds[tile * 16 + m16];
            int   cj = tile * 16 + m16;
            #pragma unroll
            for (int r = 0; r < 8; ++r) {
                float dsc = fmaf(-2.f, s[r], wn);     // relative distance
                if (dsc < best[r]) { best[r] = dsc; bidx[r] = cj; }
            }
        }
        __syncthreads();   // done reading bbuf[c&1] before it is refilled
    }

    // Butterfly min-reduce within each 16-lane half (C layout: M=r / r+8).
    // Tie-break on smaller code index to match jnp.argmin (first minimum).
    #pragma unroll
    for (int off = 8; off >= 1; off >>= 1) {
        #pragma unroll
        for (int r = 0; r < 8; ++r) {
            float ob = __shfl_xor(best[r], off, 32);
            int   oi = __shfl_xor(bidx[r], off, 32);
            if (ob < best[r] || (ob == best[r] && oi < bidx[r])) {
                best[r] = ob; bidx[r] = oi;
            }
        }
    }
    if (m16 == 0) {
        #pragma unroll
        for (int r = 0; r < 8; ++r)
            codes_lds[wave * 16 + r + laneHigh * 8] = bidx[r];
    }
    __syncthreads();

    // codes output (stored as f32 values in the concatenated output buffer)
    if (lane < 16) {
        int q = blockIdx.x * 128 + wave * 16 + lane;
        out_codes[q] = (float)codes_lds[wave * 16 + lane];
    }

    // z_q scatter (straight-through value == z_q) + deterministic loss partials
    float lsum = 0.f;
    for (int kk = 0; kk < 32; ++kk) {
        int e = kk * 32 + lane;          // 0..1023 over wave's 16 q x 64 d
        int m = e & 15;
        int d = e >> 4;
        int code = codes_lds[wave * 16 + m];
        float wq = W[code * VQ_D + d];
        int q  = blockIdx.x * 128 + wave * 16 + m;
        int oi = ((q >> 13) * VQ_D + d) * VQ_T + (q & (VQ_T - 1));
        float zv = z[oi];
        out_zq[oi] = wq;
        float df = wq - zv;
        lsum += df * df;
    }
    #pragma unroll
    for (int off = 16; off >= 1; off >>= 1) lsum += __shfl_xor(lsum, off, 32);
    if (lane == 0) wavesum[wave] = lsum;
    __syncthreads();
    if (tid == 0) {
        float s = 0.f;
        #pragma unroll
        for (int i = 0; i < 8; ++i) s += wavesum[i];
        partials[blockIdx.x] = s;
    }
}

// --- fixed-order final reduction: vq_loss = 1.25 * mean((z_q - z)^2) --------
__global__ __launch_bounds__(256)
void vq_finalize(const float* __restrict__ partials, float* __restrict__ out_loss) {
    __shared__ float sm[256];
    int tid = threadIdx.x;
    sm[tid] = partials[tid] + partials[tid + 256];
    __syncthreads();
    for (int s = 128; s > 0; s >>= 1) {
        if (tid < s) sm[tid] += sm[tid + s];
        __syncthreads();
    }
    if (tid == 0) out_loss[0] = 1.25f * sm[0] / (float)Z_ELEMS;
}

extern "C" void kernel_launch(void* const* d_in, const int* in_sizes, int n_in,
                              void* d_out, int out_size, void* d_ws, size_t ws_size,
                              hipStream_t stream) {
    const float* z = (const float*)d_in[0];          // [8,64,8192] f32
    const float* W = (const float*)d_in[1];          // [1024,64]   f32
    float* out = (float*)d_out;

    char* ws = (char*)d_ws;
    __bf16* comb    = (__bf16*)(ws);                 // 262144 B (interleaved frags)
    float*  wnorm   = (float*)(ws + 262144);         //   4096 B
    float*  partials= (float*)(ws + 266240);         //   2048 B

    float* out_zq    = out;                          // 4194304 elems
    float* out_loss  = out + Z_ELEMS;                // 1 elem
    float* out_codes = out + Z_ELEMS + 1;            // 65536 elems

    vq_prep_frags<<<512, 256, 0, stream>>>(W, comb);
    vq_prep_wnorm<<<4, 256, 0, stream>>>(W, wnorm);
    vq_main<<<512, 256, 0, stream>>>(z, W, comb, wnorm, out_zq, out_codes, partials);
    vq_finalize<<<1, 256, 0, stream>>>(partials, out_loss);
}
